// QTChessNET_76416058131305
// MI455X (gfx1250) — 8-bit matrix operand order repaired and checked by reading, compiled, not run on hardware
//
#include <hip/hip_runtime.h>
#include <stdint.h>

typedef int v8i __attribute__((ext_vector_type(8)));
typedef unsigned v4u __attribute__((ext_vector_type(4)));
typedef unsigned v8u __attribute__((ext_vector_type(8)));

constexpr size_t NPIX   = 8192ull * 64;
constexpr size_t OFF_A16 = 256;
constexpr size_t OFF_B1  = OFF_A16 + NPIX * 16;
constexpr size_t OFF_B2  = OFF_B1  + NPIX * 128;
constexpr size_t OFF_F32 = OFF_B2  + NPIX * 128;
constexpr size_t OFF_Q1  = OFF_F32 + NPIX * 128 * 4;
constexpr size_t OFF_SIG = OFF_Q1  + 8192ull * 64;
constexpr size_t OFF_W1P = OFF_SIG + 8192ull * 64 * 4;
constexpr size_t OFF_W2P = OFF_W1P + 3ull  * 8 * 32 * 32;
constexpr size_t OFF_W3P = OFF_W2P + 18ull * 8 * 32 * 32;
constexpr size_t OFF_F1P = OFF_W3P + 18ull * 8 * 32 * 32;
constexpr size_t OFF_FOP = OFF_F1P + 128ull * 4 * 32 * 32;


__device__ __forceinline__ int koffA(int v, int g) {
    return ((v & 3) << 2) + ((v >> 2) << 5) + (g << 4);
}

__global__ void k_init(unsigned* sc) {
    if (threadIdx.x < 16) sc[threadIdx.x] = 0u;
}

__global__ void k_absmax(const float* __restrict__ p, int n, unsigned* slot) {
    __shared__ float red[256];
    float m = 0.f;
    for (int i = blockIdx.x * blockDim.x + threadIdx.x; i < n; i += gridDim.x * blockDim.x)
        m = fmaxf(m, fabsf(p[i]));
    red[threadIdx.x] = m;
    __syncthreads();
    for (int s = 128; s > 0; s >>= 1) {
        if ((int)threadIdx.x < s) red[threadIdx.x] = fmaxf(red[threadIdx.x], red[threadIdx.x + s]);
        __syncthreads();
    }
    if (threadIdx.x == 0) atomicMax(slot, __float_as_uint(red[0]));
}

template <int CREAL, int CPAD, int KT>
__global__ void k_pack_conv(const float* __restrict__ w, const unsigned* __restrict__ sc,
                            int wslot, int8_t* __restrict__ dst) {
    int tid = blockIdx.x * blockDim.x + threadIdx.x;
    if (tid >= KT * 2048) return;
    int v = tid & 7, lane = (tid >> 3) & 31, mt = (tid >> 8) & 7, kt = tid >> 11;
    int g = lane >> 4, m = mt * 16 + (lane & 15);
    int kbase = kt * 64 + koffA(v, g);
    float s   = fmaxf(__uint_as_float(sc[wslot]) * (1.f / 127.f), 1e-8f);
    float inv = 1.f / s;
    int packed = 0;
#pragma unroll
    for (int j = 0; j < 4; ++j) {
        int k = kbase + j, tap = k / CPAD, c = k % CPAD;
        float wv = (tap < 9 && c < CREAL) ? w[(m * CREAL + c) * 9 + tap] : 0.f;
        int q = (int)lrintf(wv * inv);
        q = q < -128 ? -128 : (q > 127 ? 127 : q);
        packed |= (q & 0xff) << (8 * j);
    }
    ((int*)dst)[tid] = packed;
}

__global__ void k_pack_fc(const float* __restrict__ w, int K, const unsigned* __restrict__ sc,
                          int wslot, int8_t* __restrict__ dst, int total) {
    int tid = blockIdx.x * blockDim.x + threadIdx.x;
    if (tid >= total) return;
    int v = tid & 7, t = tid >> 3;
    int lane = t & 31; t >>= 5;
    int mt = t & 3, kt = t >> 2;
    int m = mt * 16 + (lane & 15);
    int kbase = kt * 64 + koffA(v, lane >> 4);
    float s   = fmaxf(__uint_as_float(sc[wslot]) * (1.f / 127.f), 1e-8f);
    float inv = 1.f / s;
    int packed = 0;
#pragma unroll
    for (int j = 0; j < 4; ++j) {
        int q = (int)lrintf(w[m * K + kbase + j] * inv);
        q = q < -128 ? -128 : (q > 127 ? 127 : q);
        packed |= (q & 0xff) << (8 * j);
    }
    ((int*)dst)[tid] = packed;
}

__global__ void k_quant_x(const float* __restrict__ x, const unsigned* __restrict__ sc,
                          int8_t* __restrict__ dst) {
    size_t t = (size_t)blockIdx.x * blockDim.x + threadIdx.x;
    if (t >= NPIX * 4) return;
    int cg = (int)(t & 3);
    size_t n = t >> 2;
    int b = (int)(n >> 6), p = (int)(n & 63);
    float s   = fmaxf(__uint_as_float(sc[0]) * (1.f / 127.f), 1e-8f);
    float inv = 1.f / s;
    int packed = 0;
#pragma unroll
    for (int j = 0; j < 4; ++j) {
        int c = cg * 4 + j;
        float xv = (c < 12) ? x[((size_t)(b * 12 + c)) * 64 + p] : 0.f;
        int q = (int)lrintf(xv * inv);
        q = q < -128 ? -128 : (q > 127 ? 127 : q);
        packed |= (q & 0xff) << (8 * j);
    }
    ((int*)dst)[n * 4 + cg] = packed;
}

__global__ void k_quant_u8(const float* __restrict__ in, const unsigned* __restrict__ sc,
                           int slot, float qmax, uint8_t* __restrict__ dst, size_t n4) {
    size_t i = (size_t)blockIdx.x * blockDim.x + threadIdx.x;
    if (i >= n4) return;
    float s   = fmaxf(__uint_as_float(sc[slot]) / qmax, 1e-8f);
    float inv = 1.f / s;
    float4 v = ((const float4*)in)[i];
    int q0 = (int)lrintf(v.x * inv), q1 = (int)lrintf(v.y * inv);
    int q2 = (int)lrintf(v.z * inv), q3 = (int)lrintf(v.w * inv);
    int qm = (int)qmax;
    q0 = q0 < 0 ? 0 : (q0 > qm ? qm : q0);
    q1 = q1 < 0 ? 0 : (q1 > qm ? qm : q1);
    q2 = q2 < 0 ? 0 : (q2 > qm ? qm : q2);
    q3 = q3 < 0 ? 0 : (q3 > qm ? qm : q3);
    ((int*)dst)[i] = q0 | (q1 << 8) | (q2 << 16) | (q3 << 24);
}

__global__ void k_quant_fc_in(const float* __restrict__ in, const unsigned* __restrict__ sc,
                              int slot, uint8_t* __restrict__ dst) {
    size_t t = (size_t)blockIdx.x * blockDim.x + threadIdx.x;
    if (t >= NPIX * 32) return;
    int cg = (int)(t & 31);
    size_t n = t >> 5;
    int b = (int)(n >> 6), p = (int)(n & 63);
    float s   = fmaxf(__uint_as_float(sc[slot]) * (1.f / 255.f), 1e-8f);
    float inv = 1.f / s;
    float4 v = ((const float4*)(in + n * 128))[cg];
    uint8_t* d = dst + (size_t)b * 8192 + (size_t)(cg * 4) * 64 + p;
    int q;
    q = (int)lrintf(v.x * inv); d[0]   = (uint8_t)(q < 0 ? 0 : (q > 255 ? 255 : q));
    q = (int)lrintf(v.y * inv); d[64]  = (uint8_t)(q < 0 ? 0 : (q > 255 ? 255 : q));
    q = (int)lrintf(v.z * inv); d[128] = (uint8_t)(q < 0 ? 0 : (q > 255 ? 255 : q));
    q = (int)lrintf(v.w * inv); d[192] = (uint8_t)(q < 0 ? 0 : (q > 255 ? 255 : q));
}

template <int CPAD>
__device__ __forceinline__ int b_lds_off(int kb, int nn) {
    int tap = kb / CPAD;
    if (tap >= 9) return 40 * CPAD;
    int c0 = kb % CPAD;
    int r  = (nn >> 3) + (tap / 3 - 1) + 1;
    int xc = (nn & 7)  + (tap % 3 - 1) + 1;
    return (r * 10 + xc) * CPAD + c0;
}

template <int CPAD, int KT, bool BSIGNED>
__global__ void __launch_bounds__(256)
k_conv(const int8_t* __restrict__ act, const int8_t* __restrict__ wpack,
       const float* __restrict__ bias, const float* __restrict__ bns,
       const float* __restrict__ bnb, const unsigned* __restrict__ sc,
       int inslot, int wslot, unsigned* amax_out, float* __restrict__ out) {
    constexpr int TILEB = 40 * CPAD;
    __shared__ __align__(16) int8_t tile[TILEB + 16];
    __shared__ float red[256];
    const int tid = threadIdx.x;
    const int n0  = blockIdx.x * 16;
    const int b   = n0 >> 6;
    const int py0 = (n0 & 63) >> 3;

    for (int i = tid; i < (TILEB + 16) / 4; i += 256) ((int*)tile)[i] = 0;
    __syncthreads();

    const int wave = tid >> 5, lane = tid & 31;
    if (py0 >= 1 && py0 <= 5) {
        constexpr unsigned PADI = (CPAD == 128) ? 7u : 4u;
        constexpr unsigned PADA = (CPAD == 128) ? 63u : 7u;
        uint64_t ga = (uint64_t)(uintptr_t)(act + ((size_t)(b * 64 + (py0 - 1) * 8)) * CPAD);
        unsigned ldsa = (unsigned)(uintptr_t)(tile + CPAD);
        v4u g0;
        g0[0] = 1u;
        g0[1] = ldsa;
        g0[2] = (unsigned)ga;
        g0[3] = (unsigned)(ga >> 32) | 0x80000000u;
        v8u g1;
        g1[0] = (1u << 20) | (PADI << 22) | (PADA << 25);
        g1[1] = ((8u * CPAD) & 0xffffu) << 16;
        g1[2] = ((8u * CPAD) >> 16) | (0xffffu << 16);
        g1[3] = 0x7fffu | ((8u * CPAD) << 16);
        g1[4] = 4u;
        g1[5] = 8u * CPAD;
        g1[6] = 0u;
        g1[7] = 0u;
        if (wave == 0) {
            asm volatile("tensor_load_to_lds %0, %1" :: "s"(g0), "s"(g1) : "memory");
        }
        __builtin_amdgcn_s_wait_tensorcnt(0);
    } else {
        constexpr int NCPR = (8 * CPAD) / 16;
        if (tid < 4 * NCPR) {
            int r = tid / NCPR, ci = tid % NCPR;
            int yy = py0 - 1 + r;
            if (yy >= 0 && yy < 8) {
                const int8_t* gsrc = act + ((size_t)(b * 64 + yy * 8)) * CPAD + ci * 16;
                unsigned ldsa = (unsigned)(uintptr_t)(tile + (r * 10 + 1) * CPAD + ci * 16);
                asm volatile("global_load_async_to_lds_b128 %0, %1, off"
                             :: "v"(ldsa), "v"(gsrc) : "memory");
            }
        }
        asm volatile("s_wait_asynccnt 0" ::: "memory");
    }
    __syncthreads();

    const int nn = lane & 15, g = lane >> 4;
    v8i acc = {};
    for (int kt = 0; kt < KT; ++kt) {
        const int8_t* wp = wpack + (((size_t)(kt * 8 + wave)) * 32 + lane) * 32;
        int4 alo = *(const int4*)wp;
        int4 ahi = *(const int4*)(wp + 16);
        v8i a;
        a[0] = alo.x; a[1] = alo.y; a[2] = alo.z; a[3] = alo.w;
        a[4] = ahi.x; a[5] = ahi.y; a[6] = ahi.z; a[7] = ahi.w;
        int kb1 = kt * 64 + g * 16;
        int4 blo = *(const int4*)(tile + b_lds_off<CPAD>(kb1, nn));
        int4 bhi = *(const int4*)(tile + b_lds_off<CPAD>(kb1 + 32, nn));
        v8i bb;
        bb[0] = blo.x; bb[1] = blo.y; bb[2] = blo.z; bb[3] = blo.w;
        bb[4] = bhi.x; bb[5] = bhi.y; bb[6] = bhi.z; bb[7] = bhi.w;
        acc = __builtin_amdgcn_wmma_i32_16x16x64_iu8(true, a, BSIGNED, bb, acc, false, false);
    }

    float sa = fmaxf(__uint_as_float(sc[inslot]) * (1.f / 127.f), 1e-8f);
    float sw = fmaxf(__uint_as_float(sc[wslot])  * (1.f / 127.f), 1e-8f);
    float st = sa * sw;
    int cbase = wave * 16 + g * 8;
    int n = n0 + nn;
    float vals[8], m = 0.f;
#pragma unroll
    for (int j = 0; j < 8; ++j) {
        int cout = cbase + j;
        float v = (float)acc[j] * st + bias[cout];
        v = v * bns[cout] + bnb[cout];
        v = fmaxf(v, 0.f);
        vals[j] = v;
        m = fmaxf(m, v);
    }
    float* op = out + (size_t)n * 128 + cbase;
    *(float4*)op       = make_float4(vals[0], vals[1], vals[2], vals[3]);
    *(float4*)(op + 4) = make_float4(vals[4], vals[5], vals[6], vals[7]);

    red[tid] = m;
    __syncthreads();
    for (int s = 128; s > 0; s >>= 1) {
        if (tid < s) red[tid] = fmaxf(red[tid], red[tid + s]);
        __syncthreads();
    }
    if (tid == 0) atomicMax(amax_out, __float_as_uint(red[0]));
}

__global__ void __launch_bounds__(128)
k_fc1(const uint8_t* __restrict__ actF, const int8_t* __restrict__ wpack,
      const float* __restrict__ bias, const unsigned* __restrict__ sc,
      int aslot, int wslot, unsigned* amax_out, float* __restrict__ out) {
    __shared__ float red[128];
    const int tid = threadIdx.x, wave = tid >> 5, lane = tid & 31;
    const int nn = lane & 15, g = lane >> 4;
    const int b0 = blockIdx.x * 16;
    const uint8_t* brow = actF + (size_t)(b0 + nn) * 8192;
    v8i acc = {};
    for (int kt = 0; kt < 128; ++kt) {
        const int8_t* wp = wpack + (((size_t)(kt * 4 + wave)) * 32 + lane) * 32;
        __builtin_prefetch(wp + 4 * 32 * 32, 0, 1);
        int4 alo = *(const int4*)wp;
        int4 ahi = *(const int4*)(wp + 16);
        v8i a;
        a[0] = alo.x; a[1] = alo.y; a[2] = alo.z; a[3] = alo.w;
        a[4] = ahi.x; a[5] = ahi.y; a[6] = ahi.z; a[7] = ahi.w;
        int4 blo = *(const int4*)(brow + kt * 64 + g * 16);
        int4 bhi = *(const int4*)(brow + kt * 64 + 32 + g * 16);
        v8i bb;
        bb[0] = blo.x; bb[1] = blo.y; bb[2] = blo.z; bb[3] = blo.w;
        bb[4] = bhi.x; bb[5] = bhi.y; bb[6] = bhi.z; bb[7] = bhi.w;
        acc = __builtin_amdgcn_wmma_i32_16x16x64_iu8(true, a, false, bb, acc, false, false);
    }
    float sa = fmaxf(__uint_as_float(sc[aslot]) * (1.f / 255.f), 1e-8f);
    float sw = fmaxf(__uint_as_float(sc[wslot]) * (1.f / 127.f), 1e-8f);
    float st = sa * sw;
    int cbase = wave * 16 + g * 8;
    int bidx = b0 + nn;
    float vals[8], m = 0.f;
#pragma unroll
    for (int j = 0; j < 8; ++j) {
        float v = fmaxf((float)acc[j] * st + bias[cbase + j], 0.f);
        vals[j] = v;
        m = fmaxf(m, v);
    }
    float* op = out + (size_t)bidx * 64 + cbase;
    *(float4*)op       = make_float4(vals[0], vals[1], vals[2], vals[3]);
    *(float4*)(op + 4) = make_float4(vals[4], vals[5], vals[6], vals[7]);
    red[tid] = m;
    __syncthreads();
    for (int s = 64; s > 0; s >>= 1) {
        if (tid < s) red[tid] = fmaxf(red[tid], red[tid + s]);
        __syncthreads();
    }
    if (tid == 0) atomicMax(amax_out, __float_as_uint(red[0]));
}

__global__ void __launch_bounds__(128)
k_fc2(const uint8_t* __restrict__ q1, const int8_t* __restrict__ wpack,
      const float* __restrict__ bias, const unsigned* __restrict__ sc,
      int aslot, int wslot, unsigned* amax_out, float* __restrict__ sig) {
    __shared__ float red[128];
    const int tid = threadIdx.x, wave = tid >> 5, lane = tid & 31;
    const int nn = lane & 15, g = lane >> 4;
    const int b0 = blockIdx.x * 16;
    const uint8_t* brow = q1 + (size_t)(b0 + nn) * 64;
    const int8_t* wp = wpack + ((size_t)wave * 32 + lane) * 32;
    int4 alo = *(const int4*)wp;
    int4 ahi = *(const int4*)(wp + 16);
    v8i a;
    a[0] = alo.x; a[1] = alo.y; a[2] = alo.z; a[3] = alo.w;
    a[4] = ahi.x; a[5] = ahi.y; a[6] = ahi.z; a[7] = ahi.w;
    int4 blo = *(const int4*)(brow + g * 16);
    int4 bhi = *(const int4*)(brow + 32 + g * 16);
    v8i bb;
    bb[0] = blo.x; bb[1] = blo.y; bb[2] = blo.z; bb[3] = blo.w;
    bb[4] = bhi.x; bb[5] = bhi.y; bb[6] = bhi.z; bb[7] = bhi.w;
    v8i acc = {};
    acc = __builtin_amdgcn_wmma_i32_16x16x64_iu8(true, a, false, bb, acc, false, false);

    float sa = fmaxf(__uint_as_float(sc[aslot]) * (1.f / 255.f), 1e-8f);
    float sw = fmaxf(__uint_as_float(sc[wslot]) * (1.f / 127.f), 1e-8f);
    float st = sa * sw;
    int cbase = wave * 16 + g * 8;
    int bidx = b0 + nn;
    float vals[8], m = 0.f;
#pragma unroll
    for (int j = 0; j < 8; ++j) {
        float v = (float)acc[j] * st + bias[cbase + j];
        v = 1.f / (1.f + expf(-v));
        vals[j] = v;
        m = fmaxf(m, v);
    }
    float* op = sig + (size_t)bidx * 64 + cbase;
    *(float4*)op       = make_float4(vals[0], vals[1], vals[2], vals[3]);
    *(float4*)(op + 4) = make_float4(vals[4], vals[5], vals[6], vals[7]);
    red[tid] = m;
    __syncthreads();
    for (int s = 64; s > 0; s >>= 1) {
        if (tid < s) red[tid] = fmaxf(red[tid], red[tid + s]);
        __syncthreads();
    }
    if (tid == 0) atomicMax(amax_out, __float_as_uint(red[0]));
}

__global__ void k_final(const float* __restrict__ sig, const unsigned* __restrict__ sc,
                        float* __restrict__ out, size_t n) {
    size_t i = (size_t)blockIdx.x * blockDim.x + threadIdx.x;
    if (i >= n) return;
    float s = fmaxf(__uint_as_float(sc[10]) * (1.f / 255.f), 1e-8f);
    int q = (int)lrintf(sig[i] / s);
    q = q < 0 ? 0 : (q > 255 ? 255 : q);
    out[i] = (float)q * s;
}

extern "C" void kernel_launch(void* const* d_in, const int* in_sizes, int n_in,
                              void* d_out, int out_size, void* d_ws, size_t ws_size,
                              hipStream_t stream) {
    const float* x     = (const float*)d_in[0];
    const float* w1    = (const float*)d_in[1];
    const float* b1    = (const float*)d_in[2];
    const float* bn1s  = (const float*)d_in[3];
    const float* bn1b  = (const float*)d_in[4];
    const float* w2    = (const float*)d_in[5];
    const float* b2    = (const float*)d_in[6];
    const float* bn2s  = (const float*)d_in[7];
    const float* bn2b  = (const float*)d_in[8];
    const float* w3    = (const float*)d_in[9];
    const float* b3    = (const float*)d_in[10];
    const float* bn3s  = (const float*)d_in[11];
    const float* bn3b  = (const float*)d_in[12];
    const float* fc1w  = (const float*)d_in[13];
    const float* fc1b  = (const float*)d_in[14];
    const float* fcow  = (const float*)d_in[15];
    const float* fcob  = (const float*)d_in[16];
    float* outp = (float*)d_out;

    uint8_t* ws = (uint8_t*)d_ws;
    unsigned* sc = (unsigned*)ws;
    int8_t*  a16  = (int8_t*)(ws + OFF_A16);
    int8_t*  buf1 = (int8_t*)(ws + OFF_B1);
    int8_t*  buf2 = (int8_t*)(ws + OFF_B2);
    float*   f32  = (float*)(ws + OFF_F32);
    uint8_t* q1   = (uint8_t*)(ws + OFF_Q1);
    float*   sig  = (float*)(ws + OFF_SIG);
    int8_t*  w1p  = (int8_t*)(ws + OFF_W1P);
    int8_t*  w2p  = (int8_t*)(ws + OFF_W2P);
    int8_t*  w3p  = (int8_t*)(ws + OFF_W3P);
    int8_t*  f1p  = (int8_t*)(ws + OFF_F1P);
    int8_t*  fop  = (int8_t*)(ws + OFF_FOP);

    k_init<<<1, 64, 0, stream>>>(sc);

    k_absmax<<<256, 256, 0, stream>>>(x, 8192 * 12 * 64, sc + 0);
    k_absmax<<<32, 256, 0, stream>>>(w1, 128 * 12 * 9, sc + 1);
    k_absmax<<<64, 256, 0, stream>>>(w2, 128 * 128 * 9, sc + 2);
    k_absmax<<<64, 256, 0, stream>>>(w3, 128 * 128 * 9, sc + 3);
    k_absmax<<<128, 256, 0, stream>>>(fc1w, 64 * 8192, sc + 4);
    k_absmax<<<4, 256, 0, stream>>>(fcow, 64 * 64, sc + 5);

    k_pack_conv<12, 16, 3><<<24, 256, 0, stream>>>(w1, sc, 1, w1p);
    k_pack_conv<128, 128, 18><<<144, 256, 0, stream>>>(w2, sc, 2, w2p);
    k_pack_conv<128, 128, 18><<<144, 256, 0, stream>>>(w3, sc, 3, w3p);
    k_pack_fc<<<512, 256, 0, stream>>>(fc1w, 8192, sc, 4, f1p, 131072);
    k_pack_fc<<<4, 256, 0, stream>>>(fcow, 64, sc, 5, fop, 1024);

    k_quant_x<<<(int)((NPIX * 4 + 255) / 256), 256, 0, stream>>>(x, sc, a16);

    k_conv<16, 3, true><<<32768, 256, 0, stream>>>(a16, w1p, b1, bn1s, bn1b, sc, 0, 1, sc + 6, f32);
    k_quant_u8<<<65536, 256, 0, stream>>>(f32, sc, 6, 127.f, (uint8_t*)buf1, NPIX * 32);
    k_conv<128, 18, false><<<32768, 256, 0, stream>>>(buf1, w2p, b2, bn2s, bn2b, sc, 6, 2, sc + 7, f32);
    k_quant_u8<<<65536, 256, 0, stream>>>(f32, sc, 7, 127.f, (uint8_t*)buf2, NPIX * 32);
    k_conv<128, 18, false><<<32768, 256, 0, stream>>>(buf2, w3p, b3, bn3s, bn3b, sc, 7, 3, sc + 8, f32);

    k_quant_fc_in<<<65536, 256, 0, stream>>>(f32, sc, 8, (uint8_t*)buf1);
    k_fc1<<<512, 128, 0, stream>>>((const uint8_t*)buf1, f1p, fc1b, sc, 8, 4, sc + 9, f32);
    k_quant_u8<<<512, 256, 0, stream>>>(f32, sc, 9, 255.f, q1, 8192 * 64 / 4);
    k_fc2<<<512, 128, 0, stream>>>(q1, fop, fcob, sc, 9, 5, sc + 10, sig);
    k_final<<<2048, 256, 0, stream>>>(sig, sc, outp, (size_t)out_size);
}
